// Recurrence_87883620810764
// MI455X (gfx1250) — matrix operand order repaired and checked by reading, not compiled, not run on hardware
//
#include <hip/hip_runtime.h>
#include <cstdint>

// ---------------------------------------------------------------------------
// Problem constants (from reference)
// ---------------------------------------------------------------------------
constexpr int H    = 256;
constexpr int S    = 64;
constexpr int AN   = 16;
constexpr int D    = 8;
constexpr int T    = 32;
constexpr int N    = 64;
constexpr int HG   = 770;          // 2 + 3*H
constexpr int HG3  = 2310;         // 3*HG
constexpr int HG3P = 2368;         // padded to multiple of 64 (N-panel)
constexpr int HGKP = 896;          // 770 padded to multiple of 128 (K-chunk)
constexpr int OUTW = 338;          // 1 + AN + 1 + H + S

// ---------------------------------------------------------------------------
// bf16 helpers / WMMA fragment types
// ---------------------------------------------------------------------------
typedef __bf16    bf16x16 __attribute__((ext_vector_type(16)));
typedef float     floatx8 __attribute__((ext_vector_type(8)));
typedef uint32_t  u32x4   __attribute__((ext_vector_type(4)));
typedef int       i32x8   __attribute__((ext_vector_type(8)));
typedef int       i32x4   __attribute__((ext_vector_type(4)));

union FragAB { bf16x16 v; uint32_t u[8]; };

__device__ __forceinline__ uint16_t f2bf(float f) {
  union { float f; uint32_t u; } x; x.f = f;
  uint32_t u = x.u;
  uint32_t r = (u + 0x7FFFu + ((u >> 16) & 1u)) >> 16;   // RNE
  return (uint16_t)r;
}

__device__ __forceinline__ float sigmoidf_(float x) { return 1.0f / (1.0f + expf(-x)); }

// ---------------------------------------------------------------------------
// TDM: 2D tile load (Global -> LDS), bf16 elements.
// D# built per CDNA5 ISA ch.8 "Tensor DMA Descriptor":
//   group0: count=1 | lds_addr | global_addr(57b) | type=2
//   group1: data_size=2B, tensor_dim0/1, tile_dim0/1, tensor_dim0_stride
//   groups 2/3: zero (2D tensor); trailing group (clang-23 6-arg form): zero.
// ---------------------------------------------------------------------------
__device__ __forceinline__ void tdm_load_2d(uint32_t lds_off_bytes,
                                            const uint16_t* gsrc,
                                            uint32_t tensor_d0, uint32_t tensor_d1,
                                            uint32_t tile_d0,   uint32_t tile_d1,
                                            uint32_t stride_elems) {
  uint64_t ga = (uint64_t)(uintptr_t)gsrc;
  u32x4 g0;
  g0[0] = 1u;                                     // count=1, user mode, no gather
  g0[1] = lds_off_bytes;                          // LDS byte address
  g0[2] = (uint32_t)ga;                           // global addr [31:0]
  g0[3] = (uint32_t)((ga >> 32) & 0x01FFFFFFu)    // global addr [56:32]
          | (2u << 30);                           // type = 2 ("image")
  i32x8 g1;
  g1[0] = (int)(1u << 16);                        // data_size = 1 (2 bytes)
  g1[1] = (int)((tensor_d0 & 0xFFFFu) << 16);     // tensor_dim0 [15:0] @ bits 63:48
  g1[2] = (int)((tensor_d0 >> 16) | ((tensor_d1 & 0xFFFFu) << 16));
  g1[3] = (int)((tensor_d1 >> 16) | (tile_d0 << 16));
  g1[4] = (int)(tile_d1 & 0xFFFFu);               // tile_dim1 (tile_dim2 = 0)
  g1[5] = (int)stride_elems;                      // tensor_dim0_stride [31:0]
  g1[6] = 0;                                      // stride hi + dim1_stride lo
  g1[7] = 0;
  i32x4 g2 = {0, 0, 0, 0};
  i32x4 g3 = {0, 0, 0, 0};
  i32x8 g4 = {0, 0, 0, 0, 0, 0, 0, 0};
  __builtin_amdgcn_tensor_load_to_lds(g0, g1, g2, g3, g4, 0);
}

// ---------------------------------------------------------------------------
// LDS-staged bf16 WMMA GEMM:  C[M x Nout] = A[M x K] * B[Nout x K]^T (+bias)(+relu)
// A, B row-major with row stride == K (K contiguous). Requirements:
//   M % 64 == 0, Nout(padded) % 64 == 0, K % 128 == 0.
// Block = 128 threads (4 waves), computes a 64x64 C tile. Per K-chunk of 128:
//   wave 0 issues two TDM tensor_load_to_lds (B panel 64x128, A panel 64x128),
//   waits TENSORcnt, workgroup barrier, then each wave runs 16 WMMAs from LDS.
// BFMODE: 0 = no bf16 mirror, 1 = bf16 mirror same layout,
//         2 = bf16 mirror in "lin" layout  bf[(m>>6)*16384 + n*64 + (m&63)]
// ---------------------------------------------------------------------------
template<int BFMODE, int RELU, int BIAS>
__global__ void __launch_bounds__(128) wmma_gemm_lds(
    const uint16_t* __restrict__ A,
    const uint16_t* __restrict__ B,
    const float*    __restrict__ bias, int bias_n,
    float*          __restrict__ C, int ldc,
    uint16_t*       __restrict__ Cbf,
    int K, int Mrows, int Nrows)
{
  __shared__ uint16_t smem[16384];   // [0,8192): B tile 64x128 ; [8192,16384): A tile

  const int tid  = threadIdx.x;
  const int w    = tid >> 5;         // wave id 0..3
  const int lane = tid & 31;
  const int hi   = lane >> 4;
  const int lo   = lane & 15;
  const int mblk = blockIdx.x * 64;
  const int nblk = blockIdx.y * 64;

  floatx8 acc[4];
#pragma unroll
  for (int i = 0; i < 4; ++i) acc[i] = (floatx8){0.f,0.f,0.f,0.f,0.f,0.f,0.f,0.f};

  for (int k0 = 0; k0 < K; k0 += 128) {
    __syncthreads();                 // previous chunk fully consumed
    if (tid < 32) {                  // wave 0 drives the Tensor Data Mover
      tdm_load_2d(0u,     B + (size_t)nblk * K + k0, (uint32_t)K, (uint32_t)Nrows,
                  128u, 64u, (uint32_t)K);
      tdm_load_2d(16384u, A + (size_t)mblk * K + k0, (uint32_t)K, (uint32_t)Mrows,
                  128u, 64u, (uint32_t)K);
      __builtin_amdgcn_s_wait_tensorcnt(0);
    }
    __syncthreads();                 // tiles visible to all waves

    const int arow = w * 16 + lo;    // this lane's A row within the tile
#pragma unroll
    for (int kk = 0; kk < 128; kk += 32) {
      FragAB a;
      // A fragment (16x32): j0..3 -> K {0..7}+hi*8, j4..7 -> K {16..23}+hi*8
      const uint16_t* ap = smem + 8192 + arow * 128 + kk + hi * 8;
      *(uint4*)&a.u[0] = *(const uint4*)ap;
      *(uint4*)&a.u[4] = *(const uint4*)(ap + 16);
#pragma unroll
      for (int nt = 0; nt < 4; ++nt) {
        FragAB b;
        // B fragment (32x16): j0..3 -> K {0..7}+hi*8, j4..7 -> K {16..23}+hi*8, col = lo
        const uint16_t* bp = smem + (nt * 16 + lo) * 128 + kk + hi * 8;
        *(uint4*)&b.u[0] = *(const uint4*)bp;
        *(uint4*)&b.u[4] = *(const uint4*)(bp + 16);
        acc[nt] = __builtin_amdgcn_wmma_f32_16x16x32_bf16(
            false, a.v, false, b.v, (short)0, acc[nt], false, false);
      }
    }
  }

  // ---- epilogue: C/D layout -> row m = mblk + w*16 + hi*8 + j, col = n ----
#pragma unroll
  for (int nt = 0; nt < 4; ++nt) {
    const int n = nblk + nt * 16 + lo;
    float bv = 0.0f;
    if (BIAS && n < bias_n) bv = bias[n];
#pragma unroll
    for (int j = 0; j < 8; ++j) {
      const int m = mblk + w * 16 + hi * 8 + j;
      float v = acc[nt][j] + bv;
      if (RELU) v = fmaxf(v, 0.0f);
      C[(size_t)m * ldc + n] = v;
      if (BFMODE == 1) {
        Cbf[(size_t)m * ldc + n] = f2bf(v);
      } else if (BFMODE == 2) {
        Cbf[((size_t)(m >> 6) << 14) + ((size_t)n << 6) + (m & 63)] = f2bf(v);
      }
    }
  }
}

// ---------------------------------------------------------------------------
// f32 -> bf16 weight conversion with K padding (zeros) and row padding (zeros)
// ---------------------------------------------------------------------------
__global__ void cvt_pad_bf16(uint16_t* __restrict__ dst,
                             const float* __restrict__ src,
                             int R, int Kc, int Kpad, int Rpad) {
  int idx = blockIdx.x * blockDim.x + threadIdx.x;
  int total = Rpad * Kpad;
  if (idx >= total) return;
  int r = idx / Kpad, k = idx - r * Kpad;
  float v = (r < R && k < Kc) ? src[(size_t)r * Kc + k] : 0.0f;
  dst[idx] = f2bf(v);
}

// Xin[(s*N + n)*H + h] = bf16(emb[subtasks[n,s], h])
__global__ void gather_emb(uint16_t* __restrict__ Xin,
                           const float* __restrict__ emb,
                           const int* __restrict__ subtasks) {
  int idx = blockIdx.x * blockDim.x + threadIdx.x;
  if (idx >= S * N * H) return;
  int h = idx & (H - 1);
  int sn = idx >> 8;           // s*N + n
  int s = sn >> 6, n = sn & 63;
  int v = subtasks[n * S + s];
  Xin[idx] = f2bf(emb[(size_t)v * H + h]);
}

__global__ void zero_h(float* __restrict__ h, uint16_t* __restrict__ hbf) {
  int idx = blockIdx.x * blockDim.x + threadIdx.x;
  if (idx >= N * HGKP) return;
  h[idx] = 0.0f; hbf[idx] = 0;
}

// GRU gate math. GI = x@Wih^T + b_ih, GH = h@Whh^T + b_hh (each [N x HG3P]).
__global__ void gru_gates(const float* __restrict__ GI, const float* __restrict__ GH,
                          float* __restrict__ h, uint16_t* __restrict__ hbf,
                          float* __restrict__ Xg, int s) {
  int idx = blockIdx.x * blockDim.x + threadIdx.x;
  if (idx >= N * HG) return;
  int n = idx / HG, g = idx - n * HG;
  const float* gi = GI + (size_t)n * HG3P;
  const float* gh = GH + (size_t)n * HG3P;
  float r  = sigmoidf_(gi[g]          + gh[g]);
  float z  = sigmoidf_(gi[HG + g]     + gh[HG + g]);
  float nn = tanhf   (gi[2 * HG + g] + r * gh[2 * HG + g]);
  float hp = h[(size_t)n * HGKP + g];
  float h2 = (1.0f - z) * nn + z * hp;
  h[(size_t)n * HGKP + g]   = h2;
  hbf[(size_t)n * HGKP + g] = f2bf(h2);
  Xg[((size_t)s * N + n) * HG + g] = h2;
}

// p_init = where(all(rnn_hxs==0), p0, rnn_hxs[:, 274:])
__global__ void p_init(float* __restrict__ p, const float* __restrict__ Xg,
                       const float* __restrict__ rnn_hxs) {
  int n = threadIdx.x;             // 64 threads
  bool allz = true;
  for (int i = 0; i < OUTW; ++i) allz = allz && (rnn_hxs[n * OUTW + i] == 0.0f);
  for (int s = 0; s < S; ++s) {
    float p0  = Xg[((size_t)s * N + n) * HG + 1];
    float hxp = rnn_hxs[n * OUTW + 1 + AN + 1 + H + s];
    p[n * S + s] = allz ? p0 : hxp;
  }
}

// Norms of Mp = X[...,514:770] and Mm = X[...,258:514], per (n,s) (t-invariant)
__global__ void mnorms(float* __restrict__ Mpn, float* __restrict__ Mmn,
                       const float* __restrict__ Xg) {
  int i = blockIdx.x * blockDim.x + threadIdx.x;
  if (i >= N * S) return;
  int n = i >> 6, s = i & 63;
  const float* row = Xg + ((size_t)s * N + n) * HG;
  float ap = 0.f, am = 0.f;
  for (int h = 0; h < H; ++h) {
    float vp = row[2 + 2 * H + h]; ap += vp * vp;
    float vm = row[2 + H + h];     am += vm * vm;
  }
  Mpn[i] = sqrtf(ap); Mmn[i] = sqrtf(am);
}

// ps = softmax(p) over S; r[n,h] = sum_s ps[n,s] * M[n,s,h]
__global__ void softmax_r(const float* __restrict__ p, float* __restrict__ ps,
                          float* __restrict__ r, const float* __restrict__ Xg) {
  int n = blockIdx.x, s = threadIdx.x;   // 64 threads
  __shared__ float buf[S];
  float pv = p[n * S + s];
  buf[s] = pv; __syncthreads();
  float mx = buf[0];
  for (int k = 1; k < S; ++k) mx = fmaxf(mx, buf[k]);
  float ev = expf(pv - mx);
  __syncthreads(); buf[s] = ev; __syncthreads();
  float sum = 0.f;
  for (int k = 0; k < S; ++k) sum += buf[k];
  float psv = ev / sum;
  ps[n * S + s] = psv;
  __syncthreads(); buf[s] = psv; __syncthreads();
  for (int h = s; h < H; h += S) {
    float acc = 0.f;
    for (int k = 0; k < S; ++k)
      acc += buf[k] * Xg[((size_t)k * N + n) * HG + 2 + h];
    r[n * H + h] = acc;
  }
}

// xbuf[(n*64+pp), h*8+d] = bf16(base[t,n,d,pp] * r[n,h])   (4096 x 2048)
__global__ void xmat(uint16_t* __restrict__ xb, const float* __restrict__ base,
                     const float* __restrict__ r, int t) {
  int idx = blockIdx.x * blockDim.x + threadIdx.x;
  if (idx >= N * 64 * H * D) return;
  int row = idx >> 11, col = idx & 2047;
  int n = row >> 6, pp = row & 63;
  int h = col >> 3, d = col & 7;
  float v = base[((((size_t)t * N + n) * D + d) << 6) + pp] * r[n * H + h];
  xb[idx] = f2bf(v);
}

// actor softmax -> masked/renormalized probs; writes out[:,0] (action) and [1:17]
__global__ void probs_k(float* __restrict__ out, const float* __restrict__ logits,
                        const float* __restrict__ interactable,
                        const int* __restrict__ actions, int t) {
  int n = threadIdx.x;               // 64 threads
  const float* lg = logits + n * 64; // logits buffer ldc = 64 (padded)
  float mx = lg[0];
  for (int j = 1; j < AN; ++j) mx = fmaxf(mx, lg[j]);
  float ex[AN]; float sum = 0.f;
  for (int j = 0; j < AN; ++j) { ex[j] = expf(lg[j] - mx); sum += ex[j]; }
  float nz[AN]; float nzsum = 0.f;
  for (int j = 0; j < AN; ++j) {
    nz[j] = (j < 5) ? 1.0f : interactable[((size_t)t * N + n) * (AN - 5) + (j - 5)];
    nzsum += nz[j];
  }
  float pr[AN]; float psum = 0.f;
  for (int j = 0; j < AN; ++j) { pr[j] = (ex[j] / sum) * nz[j]; psum += pr[j]; }
  float deficit = 1.0f - psum;
  float inv = 1.0f / fmaxf(nzsum, 1e-12f);
  float tot = 0.f;
  for (int j = 0; j < AN; ++j) {
    pr[j] += nz[j] * inv * deficit;
    pr[j] = fminf(fmaxf(pr[j], 0.0f), 1.0f);
    tot += pr[j];
  }
  float itot = 1.0f / fmaxf(tot, 1e-12f);
  float* o = out + ((size_t)t * N + n) * OUTW;
  o[0] = (float)actions[t * N + n];
  for (int j = 0; j < AN; ++j) o[1 + j] = pr[j] * itot;
}

// e[n,o] = psi_b[o] + sum_h psi_w[o, h*AN + a_n] * s[n,h]
__global__ void psi_e_k(float* __restrict__ e, const float* __restrict__ psi_w,
                        const float* __restrict__ psi_b, const float* __restrict__ s,
                        const int* __restrict__ actions, int t) {
  int idx = blockIdx.x * blockDim.x + threadIdx.x;   // 64*256
  int n = idx >> 8, o = idx & 255;
  int a = actions[t * N + n];
  const float* w = psi_w + (size_t)o * (H * AN) + a;
  const float* sv = s + n * H;
  float acc = psi_b[o];
  for (int h = 0; h < H; ++h) acc += w[h * AN] * sv[h];
  e[idx] = acc;
}

// write s to out[:,18:274]; enorm[n] = ||e[n]||; out[:,17] = s . critic_w + b
__global__ void finalize_sv(float* __restrict__ out, float* __restrict__ enorm,
                            const float* __restrict__ s, const float* __restrict__ e,
                            const float* __restrict__ critic_w,
                            const float* __restrict__ critic_b, int t) {
  int n = blockIdx.x, h = threadIdx.x;     // 256 threads
  __shared__ float red[H];
  float sv = s[n * H + h];
  out[((size_t)t * N + n) * OUTW + 1 + AN + 1 + h] = sv;
  float ev = e[n * H + h];
  red[h] = ev * ev; __syncthreads();
  for (int off = 128; off > 0; off >>= 1) {
    if (h < off) red[h] += red[h + off];
    __syncthreads();
  }
  float en = (h == 0) ? sqrtf(red[0]) : 0.f;
  __syncthreads();
  red[h] = sv * critic_w[h]; __syncthreads();
  for (int off = 128; off > 0; off >>= 1) {
    if (h < off) red[h] += red[h + off];
    __syncthreads();
  }
  if (h == 0) {
    enorm[n] = en;
    out[((size_t)t * N + n) * OUTW + 1 + AN] = red[0] + critic_b[0];
  }
}

// p_new = ps + c*cos(e,Mp) - c*cos(e,Mm); writes p and out[:,274:]
__global__ void cos_update(float* __restrict__ p, float* __restrict__ out,
                           const float* __restrict__ e, const float* __restrict__ enorm,
                           const float* __restrict__ Mpn, const float* __restrict__ Mmn,
                           const float* __restrict__ ps, const float* __restrict__ Xg,
                           int t) {
  int n = blockIdx.x, s = threadIdx.x;     // 64 threads
  const float* row = Xg + ((size_t)s * N + n) * HG;
  const float* ev = e + n * H;
  float dp = 0.f, dm = 0.f;
  for (int h = 0; h < H; ++h) {
    float eh = ev[h];
    dp += eh * row[2 + 2 * H + h];
    dm += eh * row[2 + H + h];
  }
  float en = enorm[n];
  float cp = dp / fmaxf(en * Mpn[n * S + s], 1e-8f);
  float cm = dm / fmaxf(en * Mmn[n * S + s], 1e-8f);
  float c = row[0];
  float pn = ps[n * S + s] + c * (cp - cm);
  p[n * S + s] = pn;
  out[((size_t)t * N + n) * OUTW + 1 + AN + 1 + H + s] = pn;
}

// ---------------------------------------------------------------------------
// Host side
// ---------------------------------------------------------------------------
extern "C" void kernel_launch(void* const* d_in, const int* in_sizes, int n_in,
                              void* d_out, int out_size, void* d_ws, size_t ws_size,
                              hipStream_t stream) {
  const float* base        = (const float*)d_in[0];
  const float* interactable= (const float*)d_in[1];
  const float* rnn_hxs     = (const float*)d_in[2];
  const float* emb         = (const float*)d_in[3];
  const float* gru_w_ih    = (const float*)d_in[4];
  const float* gru_w_hh    = (const float*)d_in[5];
  const float* gru_b_ih    = (const float*)d_in[6];
  const float* gru_b_hh    = (const float*)d_in[7];
  const float* conv0_w     = (const float*)d_in[8];
  const float* conv0_b     = (const float*)d_in[9];
  const float* convs_w     = (const float*)d_in[10];
  const float* convs_b     = (const float*)d_in[11];
  const float* lin_w       = (const float*)d_in[12];
  const float* lin_b       = (const float*)d_in[13];
  const float* psi_w       = (const float*)d_in[14];
  const float* psi_b       = (const float*)d_in[15];
  const float* actor_w     = (const float*)d_in[16];
  const float* actor_b     = (const float*)d_in[17];
  const float* critic_w    = (const float*)d_in[18];
  const float* critic_b    = (const float*)d_in[19];
  const int*   subtasks    = (const int*)d_in[20];
  const int*   actions     = (const int*)d_in[21];
  float* out = (float*)d_out;

  // ---- workspace bump allocator (all buffers fully rewritten every call) ----
  size_t off = 0;
  auto alloc = [&](size_t bytes) -> void* {
    bytes = (bytes + 255) & ~(size_t)255;
    void* p = (char*)d_ws + off; off += bytes; return p;
  };
  uint16_t* Wih_bf  = (uint16_t*)alloc((size_t)HG3P * 256 * 2);
  uint16_t* Whh_bf  = (uint16_t*)alloc((size_t)HG3P * HGKP * 2);
  uint16_t* W0_bf   = (uint16_t*)alloc((size_t)256 * 2048 * 2);
  uint16_t* W1_bf   = (uint16_t*)alloc((size_t)256 * 256 * 2);
  uint16_t* W2_bf   = (uint16_t*)alloc((size_t)256 * 256 * 2);
  uint16_t* Wlin_bf = (uint16_t*)alloc((size_t)256 * 16384 * 2);
  uint16_t* Wact_bf = (uint16_t*)alloc((size_t)64 * 256 * 2);
  uint16_t* Xin_bf  = (uint16_t*)alloc((size_t)S * N * H * 2);
  float*    GI      = (float*)alloc((size_t)N * HG3P * 4);
  float*    GH      = (float*)alloc((size_t)N * HG3P * 4);
  float*    hbuf    = (float*)alloc((size_t)N * HGKP * 4);
  uint16_t* hbf     = (uint16_t*)alloc((size_t)N * HGKP * 2);
  float*    Xg      = (float*)alloc((size_t)S * N * HG * 4);
  float*    pbuf    = (float*)alloc((size_t)N * S * 4);
  float*    psbuf   = (float*)alloc((size_t)N * S * 4);
  float*    Mpn     = (float*)alloc((size_t)N * S * 4);
  float*    Mmn     = (float*)alloc((size_t)N * S * 4);
  float*    rbuf    = (float*)alloc((size_t)N * H * 4);
  uint16_t* xbuf    = (uint16_t*)alloc((size_t)4096 * 2048 * 2);
  float*    c0f     = (float*)alloc((size_t)4096 * 256 * 4);
  uint16_t* c0bf    = (uint16_t*)alloc((size_t)4096 * 256 * 2);
  float*    c1f     = (float*)alloc((size_t)4096 * 256 * 4);
  uint16_t* c1bf    = (uint16_t*)alloc((size_t)4096 * 256 * 2);
  float*    c2f     = (float*)alloc((size_t)4096 * 256 * 4);
  uint16_t* c2lin   = (uint16_t*)alloc((size_t)64 * 16384 * 2);
  float*    sbuf    = (float*)alloc((size_t)N * H * 4);
  uint16_t* sbf     = (uint16_t*)alloc((size_t)N * H * 2);
  float*    logits  = (float*)alloc((size_t)64 * 64 * 4);
  float*    ebuf    = (float*)alloc((size_t)N * H * 4);
  float*    enormb  = (float*)alloc((size_t)N * 4);
  (void)ws_size; (void)n_in; (void)in_sizes; (void)out_size;

  auto blocks = [](long total, int bs) { return (int)((total + bs - 1) / bs); };

  // ---- weight conversion (f32 -> bf16, zero-padded) ----
  cvt_pad_bf16<<<blocks((long)HG3P * 256, 256), 256, 0, stream>>>(Wih_bf, gru_w_ih, HG3, 256, 256, HG3P);
  cvt_pad_bf16<<<blocks((long)HG3P * HGKP, 256), 256, 0, stream>>>(Whh_bf, gru_w_hh, HG3, HG, HGKP, HG3P);
  cvt_pad_bf16<<<blocks(256L * 2048, 256), 256, 0, stream>>>(W0_bf, conv0_w, 256, 2048, 2048, 256);
  cvt_pad_bf16<<<blocks(256L * 256, 256), 256, 0, stream>>>(W1_bf, convs_w, 256, 256, 256, 256);
  cvt_pad_bf16<<<blocks(256L * 256, 256), 256, 0, stream>>>(W2_bf, convs_w + 256 * 256, 256, 256, 256, 256);
  cvt_pad_bf16<<<blocks(256L * 16384, 256), 256, 0, stream>>>(Wlin_bf, lin_w, 256, 16384, 16384, 256);
  cvt_pad_bf16<<<blocks(64L * 256, 256), 256, 0, stream>>>(Wact_bf, actor_w, 16, 256, 256, 64);
  gather_emb<<<blocks((long)S * N * H, 256), 256, 0, stream>>>(Xin_bf, emb, subtasks);
  zero_h<<<blocks((long)N * HGKP, 256), 256, 0, stream>>>(hbuf, hbf);

  // ---- GRU: 64 sequential steps, each = 2 TDM+WMMA GEMMs + gate kernel ----
  for (int s = 0; s < S; ++s) {
    wmma_gemm_lds<0, 0, 1><<<dim3(1, HG3P / 64), 128, 0, stream>>>(
        Xin_bf + (size_t)s * N * H, Wih_bf, gru_b_ih, HG3, GI, HG3P, nullptr,
        H, N, HG3P);
    wmma_gemm_lds<0, 0, 1><<<dim3(1, HG3P / 64), 128, 0, stream>>>(
        hbf, Whh_bf, gru_b_hh, HG3, GH, HG3P, nullptr,
        HGKP, N, HG3P);
    gru_gates<<<blocks((long)N * HG, 256), 256, 0, stream>>>(GI, GH, hbuf, hbf, Xg, s);
  }

  p_init<<<1, 64, 0, stream>>>(pbuf, Xg, rnn_hxs);
  mnorms<<<blocks((long)N * S, 256), 256, 0, stream>>>(Mpn, Mmn, Xg);

  // ---- scan over T timesteps ----
  for (int t = 0; t < T; ++t) {
    softmax_r<<<N, S, 0, stream>>>(pbuf, psbuf, rbuf, Xg);
    xmat<<<blocks((long)4096 * 2048, 256), 256, 0, stream>>>(xbuf, base, rbuf, t);
    // conv0: (4096 x 2048) @ (2048 x 256)^T, relu, bf16 mirror
    wmma_gemm_lds<1, 1, 1><<<dim3(4096 / 64, 256 / 64), 128, 0, stream>>>(
        xbuf, W0_bf, conv0_b, 256, c0f, 256, c0bf, 2048, 4096, 256);
    // conv1
    wmma_gemm_lds<1, 1, 1><<<dim3(4096 / 64, 256 / 64), 128, 0, stream>>>(
        c0bf, W1_bf, convs_b, 256, c1f, 256, c1bf, 256, 4096, 256);
    // conv2 -> bf16 in lin layout [n][o*64 + p]
    wmma_gemm_lds<2, 1, 1><<<dim3(4096 / 64, 256 / 64), 128, 0, stream>>>(
        c1bf, W2_bf, convs_b + 256, 256, c2f, 256, c2lin, 256, 4096, 256);
    // lin: (64 x 16384) @ (16384 x 256)^T, relu, bf16 mirror -> s
    wmma_gemm_lds<1, 1, 1><<<dim3(1, 256 / 64), 128, 0, stream>>>(
        c2lin, Wlin_bf, lin_b, 256, sbuf, 256, sbf, 16384, 64, 256);
    // actor logits: (64 x 256) @ (256 x 16)^T  (N padded to 64, ldc 64)
    wmma_gemm_lds<0, 0, 1><<<dim3(1, 1), 128, 0, stream>>>(
        sbf, Wact_bf, actor_b, 16, logits, 64, nullptr, 256, 64, 64);
    probs_k<<<1, 64, 0, stream>>>(out, logits, interactable, actions, t);
    psi_e_k<<<blocks((long)N * H, 256), 256, 0, stream>>>(ebuf, psi_w, psi_b, sbuf, actions, t);
    finalize_sv<<<N, H, 0, stream>>>(out, enormb, sbuf, ebuf, critic_w, critic_b, t);
    cos_update<<<N, S, 0, stream>>>(pbuf, out, ebuf, enormb, Mpn, Mmn, psbuf, Xg, t);
  }

  // second output = last timestep, appended after hx_out
  (void)hipMemcpyAsync(out + (size_t)T * N * OUTW, out + (size_t)(T - 1) * N * OUTW,
                       (size_t)N * OUTW * sizeof(float), hipMemcpyDeviceToDevice, stream);
}
